// OtherTreeLSTM_26087631356091
// MI455X (gfx1250) — compile-verified, matrix operand order checked by reading, not run on hardware
//
#include <hip/hip_runtime.h>
#include <hip/hip_bf16.h>
#include <math.h>

// ---------------- problem constants ----------------
constexpr int kB  = 8;
constexpr int kN  = 16383;   // nodes
constexpr int kIN = 300;
constexpr int kH  = 512;
constexpr int kE  = 100;
constexpr int kDepth = 14;   // log2(16384)
constexpr int kH3 = 3 * kH;  // 1536
// padded K dims (multiple of 32 -> clean WMMA K tiles, 16B-aligned bf16 rows)
constexpr int kINp = 320;
constexpr int kEp  = 128;
// LDS B-panel row pitch (ushorts): 32 + 8 pad -> conflict-free b128 fragment reads
constexpr int kPitch = 40;

typedef __attribute__((ext_vector_type(16))) __bf16          v16bf;
typedef __attribute__((ext_vector_type(16))) unsigned short  v16u;
typedef __attribute__((ext_vector_type(8)))  unsigned short  v8u;
typedef __attribute__((ext_vector_type(8)))  float           v8f;

__device__ __forceinline__ unsigned short f2bf(float f) {
  union { float f; unsigned u; } x; x.f = f;
  unsigned u = x.u;
  u += 0x7fffu + ((u >> 16) & 1u);   // round-to-nearest-even
  return (unsigned short)(u >> 16);
}
__device__ __forceinline__ float sigf(float x) { return 1.0f / (1.0f + __expf(-x)); }

enum { EPI_STORE = 0, EPI_ACCUM = 1, EPI_GATE_BF16 = 2, EPI_GATE_F32 = 3 };

// C = A * W^T, bf16 WMMA with f32 accumulate.
//   A: bf16, contiguous rows (M x lda), zero-padded to lda
//   W: bf16, row-major (O x ldw), zero-padded to ldw
// Block = 8 waves (256 threads), tile = 128 rows x 128 cols.
// The 128x32 B chunk per K-step is cooperatively staged into double-buffered LDS
// and shared by all 8 waves (8x reduction of global B traffic); each wave computes
// a 16x128 slice with 8 accumulators and one A fragment per K-step.
template <int MODE>
__global__ __launch_bounds__(256)
void wmma_gemm_kernel(const unsigned short* __restrict__ A, int lda,
                      const unsigned short* __restrict__ W, int ldw,
                      int M, int N, int Kp,
                      const float* __restrict__ bias, const float* __restrict__ bias2,
                      float* __restrict__ C, int ldc,
                      const float* __restrict__ extra, long eBatch, int eRow, int nPerB,
                      const float* __restrict__ fxAdd,
                      void* __restrict__ outSum, int ldo)
{
  __shared__ unsigned short smem[2][128 * kPitch];   // 2 x 5120 ushorts = 20 KB

  const int tid     = threadIdx.x;
  const int wave    = tid >> 5;
  const int lane    = tid & 31;
  const int rowBase = blockIdx.x * 128 + wave * 16;
  const int colBase = blockIdx.y * 128;
  const int lrow    = lane & 15;
  const int kOff    = (lane >> 4) << 3;   // 0 or 8 (A/B K striping per ISA layout)
  const int m0      = (lane >> 4) << 3;   // 0 or 8 (C/D row striping)

  int ar = rowBase + lrow;
  if (ar >= M) ar = M - 1;                // clamp (reads valid rows, stores masked)
  const unsigned short* aP = A + (long)ar * lda + kOff;

  // staging assignment: thread -> (W row w, 16-elem half) of the 128x32 chunk
  const int sw   = tid >> 1;              // 0..127
  const int shalf = (tid & 1) * 16;       // 0 or 16
  const unsigned short* wSrc = W + (long)(colBase + sw) * ldw + shalf;
  unsigned short* wDst = &smem[0][sw * kPitch + shalf];
  unsigned short* wDst1 = &smem[1][sw * kPitch + shalf];

  const v8f zero = {0.f, 0.f, 0.f, 0.f, 0.f, 0.f, 0.f, 0.f};
  v8f acc[8] = {zero, zero, zero, zero, zero, zero, zero, zero};

  const int nk = Kp >> 5;

  // stage first chunk
  {
    const v8u* s = reinterpret_cast<const v8u*>(wSrc);
    v8u x0 = s[0], x1 = s[1];
    v8u* d = reinterpret_cast<v8u*>(wDst);
    d[0] = x0; d[1] = x1;
  }
  __syncthreads();

  for (int s = 0; s < nk; ++s) {
    const int cur = s & 1;
    // stage next chunk into the other buffer (overlaps with WMMA below)
    if (s + 1 < nk) {
      const v8u* sp = reinterpret_cast<const v8u*>(wSrc + (long)(s + 1) * 32);
      v8u x0 = sp[0], x1 = sp[1];
      v8u* d = reinterpret_cast<v8u*>(cur ? wDst : wDst1);
      d[0] = x0; d[1] = x1;
    }
    // A fragment (per-wave, streamed from global)
    const int kb = s * 32;
    v8u a0 = *reinterpret_cast<const v8u*>(aP + kb);
    v8u a1 = *reinterpret_cast<const v8u*>(aP + kb + 16);
    v16u au;
#pragma unroll
    for (int i = 0; i < 8; ++i) { au[i] = a0[i]; au[i + 8] = a1[i]; }
    v16bf a = __builtin_bit_cast(v16bf, au);
    // 8 WMMAs from LDS B panel
    const unsigned short* sm = smem[cur];
#pragma unroll
    for (int t = 0; t < 8; ++t) {
      v8u w0 = *reinterpret_cast<const v8u*>(&sm[(t * 16 + lrow) * kPitch + kOff]);
      v8u w1 = *reinterpret_cast<const v8u*>(&sm[(t * 16 + lrow) * kPitch + 16 + kOff]);
      v16u bu;
#pragma unroll
      for (int i = 0; i < 8; ++i) { bu[i] = w0[i]; bu[i + 8] = w1[i]; }
      v16bf b = __builtin_bit_cast(v16bf, bu);
      acc[t] = __builtin_amdgcn_wmma_f32_16x16x32_bf16(
          /*neg_a=*/false, a, /*neg_b=*/false, b,
          /*c_mod=*/(short)0, acc[t], /*reuse_a=*/false, /*reuse_b=*/false);
    }
    __syncthreads();   // cur reads done + next stage visible
  }

  // ---- epilogue: acc VGPR v, lane l -> row = rowBase + v + m0, col = colBase + t*16 + (l&15)
  if (MODE == EPI_STORE || MODE == EPI_ACCUM) {
#pragma unroll
    for (int t = 0; t < 8; ++t) {
      int col = colBase + t * 16 + lrow;
      float bv = 0.f;
      if (MODE == EPI_STORE) {
        if (bias)  bv += bias[col];
        if (bias2) bv += bias2[col];
      }
#pragma unroll
      for (int v = 0; v < 8; ++v) {
        int row = rowBase + m0 + v;
        if (row < M) {
          long off = (long)row * ldc + col;
          if (MODE == EPI_ACCUM) C[off] += acc[t][v];
          else                   C[off]  = acc[t][v] + bv;
        }
      }
    }
  } else {  // gate + child-pair weighted sum (rows 2j, 2j+1 adjacent in acc VGPRs)
#pragma unroll
    for (int t = 0; t < 8; ++t) {
      int col = colBase + t * 16 + lrow;
      float bv = bias ? bias[col] : 0.f;
#pragma unroll
      for (int v = 0; v < 8; v += 2) {
        int r0 = rowBase + m0 + v;
        if (r0 + 1 < M) {
          int p = r0 >> 1;  // parent row (children adjacent, batch-aligned)
          float fxv = fxAdd ? fxAdd[(long)p * kH + col] : 0.f;
          const float* e0 = extra + (long)(r0 / nPerB) * eBatch + (long)(r0 % nPerB) * eRow;
          const float* e1 = extra + (long)((r0 + 1) / nPerB) * eBatch + (long)((r0 + 1) % nPerB) * eRow;
          float g0 = sigf(acc[t][v]     + bv + fxv);
          float g1 = sigf(acc[t][v + 1] + bv + fxv);
          float s2 = g0 * e0[col] + g1 * e1[col];
          if (MODE == EPI_GATE_BF16)
            ((unsigned short*)outSum)[(long)p * ldo + col] = f2bf(s2);
          else
            ((float*)outSum)[(long)p * ldo + col] = s2;
        }
      }
    }
  }
}

// Gather a (B,N,*) tree view into contiguous zero-padded bf16 rows (rows x Kp).
__global__ void gather_bf16_kernel(const float* __restrict__ src, long sBatch, int sRow, int nPerB,
                                   unsigned short* __restrict__ dst, int K, int Kp, long total) {
  long idx = (long)blockIdx.x * blockDim.x + threadIdx.x;   // over rows*Kp
  if (idx >= total) return;
  long r = idx / Kp;
  int  k = (int)(idx % Kp);
  unsigned short v = 0;
  if (k < K)
    v = f2bf(src[(r / nPerB) * sBatch + (long)(r % nPerB) * sRow + k]);
  dst[idx] = v;
}

// f32 -> bf16 weight conversion with zero K-padding (tiny; weights live in L2)
__global__ void convert_weights_kernel(const float* __restrict__ W, unsigned short* __restrict__ Wb,
                                       int O, int K, int ldw) {
  long idx = (long)blockIdx.x * blockDim.x + threadIdx.x;
  long total = (long)O * ldw;
  if (idx >= total) return;
  int r = (int)(idx / ldw), k = (int)(idx % ldw);
  Wb[idx] = (k < K) ? f2bf(W[(long)r * K + k]) : (unsigned short)0;
}

// LSTM cell: c = sig(i)*tanh(u) + fc_sum ; h = sig(o)*tanh(c); scatter into tree buffers
__global__ void cell_kernel(const float* __restrict__ iou, const float* __restrict__ fcsum,
                            float* __restrict__ cT, float* __restrict__ hT,
                            int n, int start, long total) {
  long idx = (long)blockIdx.x * blockDim.x + threadIdx.x;
  if (idx >= total) return;
  int col = (int)(idx % kH);
  long p  = idx / kH;
  int b    = (int)(p / n);
  int node = (int)(p % n);
  const float* r = iou + p * (long)kH3;
  float iv = r[col], ov = r[kH + col], uv = r[2 * kH + col];
  float fc = fcsum ? fcsum[p * (long)kH + col] : 0.f;
  float cv = sigf(iv) * tanhf(uv) + fc;
  float hv = sigf(ov) * tanhf(cv);
  long off = ((long)b * kN + start + node) * (long)kH + col;
  cT[off] = cv;
  hT[off] = hv;
}

// out = concat(c[:,0], h[:,0])  -> (B, 2H)
__global__ void out_kernel(const float* __restrict__ cT, const float* __restrict__ hT,
                           float* __restrict__ out) {
  int idx = blockIdx.x * blockDim.x + threadIdx.x;
  if (idx >= kB * 2 * kH) return;
  int b = idx / (2 * kH), col = idx % (2 * kH);
  long base = (long)b * kN * kH;
  out[idx] = (col < kH) ? cT[base + col] : hT[base + col - kH];
}

extern "C" void kernel_launch(void* const* d_in, const int* in_sizes, int n_in,
                              void* d_out, int out_size, void* d_ws, size_t ws_size,
                              hipStream_t stream) {
  (void)in_sizes; (void)n_in; (void)out_size; (void)ws_size;
  const float* inputs = (const float*)d_in[0];   // (B,N,IN)
  const float* edge   = (const float*)d_in[1];   // (B,N,E)
  const float* Wioux  = (const float*)d_in[2];   // (3H,IN)
  const float* bioux  = (const float*)d_in[3];
  const float* Wiouh  = (const float*)d_in[4];   // (3H,H)
  const float* biouh  = (const float*)d_in[5];
  const float* Wfx    = (const float*)d_in[6];   // (H,IN)
  const float* bfx    = (const float*)d_in[7];
  const float* Wfh    = (const float*)d_in[8];   // (H,H)
  const float* bfh    = (const float*)d_in[9];
  const float* We     = (const float*)d_in[10];  // (H,E)
  const float* be     = (const float*)d_in[11];
  float* out = (float*)d_out;

  // ---- carve workspace ----
  char* ws = (char*)d_ws;
  size_t off = 0;
  auto alloc = [&](size_t bytes) -> void* {
    void* p = ws + off;
    off = (off + bytes + 255) & ~(size_t)255;
    return p;
  };
  float* hT    = (float*)alloc((size_t)kB * kN * kH * 4);
  float* cT    = (float*)alloc((size_t)kB * kN * kH * 4);
  float* iou   = (float*)alloc((size_t)kB * 8192 * kH3 * 4);
  float* fx    = (float*)alloc((size_t)kB * 4096 * kH * 4);
  float* fcs   = (float*)alloc((size_t)kB * 4096 * kH * 4);
  unsigned short* xb    = (unsigned short*)alloc((size_t)kB * 8192 * kINp * 2);  // x, bf16
  unsigned short* wcb   = (unsigned short*)alloc((size_t)kB * 8192 * kEp * 2);   // wc (child rows)
  unsigned short* hcb   = (unsigned short*)alloc((size_t)kB * 8192 * kH * 2);    // hc (child rows)
  unsigned short* hsumb = (unsigned short*)alloc((size_t)kB * 4096 * kH * 2);    // h_sum, bf16
  unsigned short* WiouxB = (unsigned short*)alloc((size_t)kH3 * kINp * 2);
  unsigned short* WiouhB = (unsigned short*)alloc((size_t)kH3 * kH * 2);
  unsigned short* WfxB   = (unsigned short*)alloc((size_t)kH * kINp * 2);
  unsigned short* WfhB   = (unsigned short*)alloc((size_t)kH * kH * 2);
  unsigned short* WeB    = (unsigned short*)alloc((size_t)kH * kEp * 2);

  // ---- convert weights to padded bf16 ----
  auto cvt = [&](const float* Wp, unsigned short* Wb, int O, int K, int ldw) {
    long total = (long)O * ldw;
    convert_weights_kernel<<<dim3((unsigned)((total + 255) / 256)), 256, 0, stream>>>(Wp, Wb, O, K, ldw);
  };
  cvt(Wioux, WiouxB, kH3, kIN, kINp);
  cvt(Wiouh, WiouhB, kH3, kH,  kH);
  cvt(Wfx,   WfxB,   kH,  kIN, kINp);
  cvt(Wfh,   WfhB,   kH,  kH,  kH);
  cvt(We,    WeB,    kH,  kE,  kEp);

  auto gather = [&](const float* src, long sBatch, int sRow, int nPerB,
                    unsigned short* dst, int K, int Kp, long rows) {
    long total = rows * (long)Kp;
    gather_bf16_kernel<<<dim3((unsigned)((total + 255) / 256)), 256, 0, stream>>>(
        src, sBatch, sRow, nPerB, dst, K, Kp, total);
  };

  // ---- bottom-up level sweep (stream order carries the tree dependency) ----
  for (int lvl = kDepth - 1; lvl >= 0; --lvl) {
    const int startN = (1 << lvl) - 1;
    const int n      = 1 << lvl;
    const int rows   = kB * n;
    const bool leaf  = (lvl == kDepth - 1);
    const int cs     = 2 * startN + 1;
    const int rows2  = 2 * rows;

    // x -> bf16 (consumed by iou and fx GEMMs)
    gather(inputs + (long)startN * kIN, (long)kN * kIN, kIN, n, xb, kIN, kINp, rows);

    if (!leaf) {
      // wc, hc child rows -> bf16
      gather(edge + (long)cs * kE, (long)kN * kE, kE, 2 * n, wcb, kE, kEp, rows2);
      gather(hT + (long)cs * kH, (long)kN * kH, kH, 2 * n, hcb, kH, kH, rows2);

      // ex = sigmoid(wc·We^T + be); h_sum = pair-sum(ex * hc) -> bf16   [fused epilogue]
      dim3 g1((unsigned)((rows2 + 127) / 128), kH / 128);
      wmma_gemm_kernel<EPI_GATE_BF16><<<g1, 256, 0, stream>>>(
          wcb, kEp, WeB, kEp, rows2, kH, kEp,
          be, nullptr, nullptr, 0,
          hT + (long)cs * kH, (long)kN * kH, kH, 2 * n,
          nullptr, hsumb, kH);
    }

    // iou = x·Wioux^T + bioux + biouh
    dim3 g2((unsigned)((rows + 127) / 128), kH3 / 128);
    wmma_gemm_kernel<EPI_STORE><<<g2, 256, 0, stream>>>(
        xb, kINp, WiouxB, kINp, rows, kH3, kINp,
        bioux, biouh, iou, kH3,
        nullptr, 0, 0, 1, nullptr, nullptr, 0);

    if (!leaf) {
      // iou += h_sum·Wiouh^T
      wmma_gemm_kernel<EPI_ACCUM><<<g2, 256, 0, stream>>>(
          hsumb, kH, WiouhB, kH, rows, kH3, kH,
          nullptr, nullptr, iou, kH3,
          nullptr, 0, 0, 1, nullptr, nullptr, 0);
      // fx = x·Wfx^T + bfx
      dim3 g3((unsigned)((rows + 127) / 128), kH / 128);
      wmma_gemm_kernel<EPI_STORE><<<g3, 256, 0, stream>>>(
          xb, kINp, WfxB, kINp, rows, kH, kINp,
          bfx, nullptr, fx, kH,
          nullptr, 0, 0, 1, nullptr, nullptr, 0);
      // f = sigmoid(hc·Wfh^T + bfh + fx); fc_sum = pair-sum(f * cc) -> f32   [fused epilogue]
      dim3 g4((unsigned)((rows2 + 127) / 128), kH / 128);
      wmma_gemm_kernel<EPI_GATE_F32><<<g4, 256, 0, stream>>>(
          hcb, kH, WfhB, kH, rows2, kH, kH,
          bfh, nullptr, nullptr, 0,
          cT + (long)cs * kH, (long)kN * kH, kH, 2 * n,
          fx, fcs, kH);
    }

    // cell update into tree buffers
    long total = (long)rows * kH;
    cell_kernel<<<dim3((unsigned)((total + 255) / 256)), 256, 0, stream>>>(
        iou, leaf ? nullptr : fcs, cT, hT, n, startN, total);
  }

  out_kernel<<<dim3((kB * 2 * kH + 255) / 256), 256, 0, stream>>>(cT, hT, out);
}
